// LaneTrajClassifier_29970281791545
// MI455X (gfx1250) — matrix operand order repaired and checked by reading, not compiled, not run on hardware
//
#include <hip/hip_runtime.h>
#include <hip/hip_bf16.h>

// LaneTrajClassifier for MI455X (gfx1250, wave32, WMMA f32_16x16x32_f16).
// All matmul dims are padded so every WMMA fragment load is unconditional:
// A / B^T fragments load as 4x float4 (global_load_b128), weight fragments as
// lane-coalesced b32 loads. fp32 accumulate; f32->f16 conversion in-register.
// Agent dim 33 is padded to 64 (zero rows / masked softmax columns).
// GEMM epilogue variants are templated (act / residual mode) so no runtime
// branches or integer division appear in the store path.
// Workspace: ~95.2M floats (~381 MB) with traj-decoder buffers aliased onto
// dead early regions.

typedef __attribute__((ext_vector_type(16))) _Float16 v16h;
typedef __attribute__((ext_vector_type(8)))  float    v8f;

// ---------------------------------------------------------------------------
// WMMA fragment loaders (wave32, fully aligned, no bounds checks).
// A 16x32 f16 layout: lanes 0-15 M=lane (K: e -> klo+e, klo=0),
// lanes 16-31 M=lane-16, klo=16.  Each half is one contiguous 16-float
// run -> 4x float4 loads per lane.
// B (KxN): lane holds column n=lane&15; element e -> K = kb + e, kb = 16*(lane>=16).
// C/D f32: vgpr r, lanes 0-15 -> (M=r, N=lane); lanes 16-31 -> (M=r+8, N=lane-16).
// ---------------------------------------------------------------------------
__device__ __forceinline__ v16h load_a_frag(const float* __restrict__ A, long long ldA,
                                            int mBase, int k0, int lane) {
  const float* row = A + (long long)(mBase + (lane & 15)) * ldA + k0 + ((lane & 16) ? 16 : 0);
  float4 a0 = ((const float4*)row)[0];
  float4 a1 = ((const float4*)row)[1];
  float4 a2 = ((const float4*)row)[2];
  float4 a3 = ((const float4*)row)[3];
  v16h f;
  f[0]  = (_Float16)a0.x; f[1]  = (_Float16)a0.y; f[2]  = (_Float16)a0.z; f[3]  = (_Float16)a0.w;
  f[4]  = (_Float16)a1.x; f[5]  = (_Float16)a1.y; f[6]  = (_Float16)a1.z; f[7]  = (_Float16)a1.w;
  f[8]  = (_Float16)a2.x; f[9]  = (_Float16)a2.y; f[10] = (_Float16)a2.z; f[11] = (_Float16)a2.w;
  f[12] = (_Float16)a3.x; f[13] = (_Float16)a3.y; f[14] = (_Float16)a3.z; f[15] = (_Float16)a3.w;
  return f;
}

// B row-major (Kdim x N): column n strided by N; loads coalesce across lanes.
__device__ __forceinline__ v16h load_b_frag(const float* __restrict__ B, int N,
                                            int k0, int nBase, int lane) {
  const float* col = B + (long long)(k0 + ((lane & 16) ? 16 : 0)) * N + nBase + (lane & 15);
  v16h f;
#pragma unroll
  for (int e = 0; e < 16; ++e) f[e] = (_Float16)col[(long long)e * N];
  return f;
}

// Transposed B: memory is (N x Kdim) row-major; logical B[k][n] = mem[n*ldB + k].
// Contiguous 16 floats per lane -> 4x float4.
__device__ __forceinline__ v16h load_bt_frag(const float* __restrict__ B, long long ldB,
                                             int k0, int nBase, int lane) {
  const float* row = B + (long long)(nBase + (lane & 15)) * ldB + k0 + ((lane & 16) ? 16 : 0);
  float4 a0 = ((const float4*)row)[0];
  float4 a1 = ((const float4*)row)[1];
  float4 a2 = ((const float4*)row)[2];
  float4 a3 = ((const float4*)row)[3];
  v16h f;
  f[0]  = (_Float16)a0.x; f[1]  = (_Float16)a0.y; f[2]  = (_Float16)a0.z; f[3]  = (_Float16)a0.w;
  f[4]  = (_Float16)a1.x; f[5]  = (_Float16)a1.y; f[6]  = (_Float16)a1.z; f[7]  = (_Float16)a1.w;
  f[8]  = (_Float16)a2.x; f[9]  = (_Float16)a2.y; f[10] = (_Float16)a2.z; f[11] = (_Float16)a2.w;
  f[12] = (_Float16)a3.x; f[13] = (_Float16)a3.y; f[14] = (_Float16)a3.z; f[15] = (_Float16)a3.w;
  return f;
}

// ---------------------------------------------------------------------------
// Generic GEMM: C = act(A @ W + bias) + res.  All dims aligned (M%16==0,
// N%16==0, K%32==0).  One wave per 16x16 tile; grid = (N/16, M/16).
// ACT: 0 = identity, 1 = leaky-relu(0.1).
// RESMODE: 0 = none, 1 = flat res[m*N+n],
//          2 = broadcast over L: res row = (m/512)*64 + (m&63)  (rowsPer=64, div=8)
// ---------------------------------------------------------------------------
template <int ACT, int RESMODE>
__global__ __launch_bounds__(32)
void gemm_kernel(const float* __restrict__ A, const float* __restrict__ W,
                 const float* __restrict__ bias, const float* __restrict__ res,
                 float* __restrict__ C, int N, int Kdim) {
  int nBase = blockIdx.x << 4;
  int mBase = blockIdx.y << 4;
  int lane = threadIdx.x;
  v8f acc = {};
  for (int k0 = 0; k0 < Kdim; k0 += 32) {
    v16h af = load_a_frag(A, Kdim, mBase, k0, lane);
    v16h bf = load_b_frag(W, N, k0, nBase, lane);
    acc = __builtin_amdgcn_wmma_f32_16x16x32_f16(false, af, false, bf,
                                                 (short)0, acc, false, false);
  }
  int n = nBase + (lane & 15);
  int mOff = (lane & 16) ? 8 : 0;
  float bv = bias ? bias[n] : 0.0f;

  float rv[8];
  if (RESMODE != 0) {
#pragma unroll
    for (int r = 0; r < 8; ++r) {
      int m = mBase + r + mOff;
      long long ri;
      if (RESMODE == 1) ri = (long long)m * N + n;                       // flat
      else              ri = (long long)(((m >> 9) << 6) | (m & 63)) * N + n; // bcast L
      rv[r] = res[ri];
    }
  }
#pragma unroll
  for (int r = 0; r < 8; ++r) {
    int m = mBase + r + mOff;
    float v = acc[r] + bv;
    if (ACT) v = (v > 0.0f) ? v : 0.1f * v;   // leaky relu, NEG_SLOPE = 0.1
    if (RESMODE != 0) v += rv[r];
    C[(long long)m * N + n] = v;
  }
}

// ---------------------------------------------------------------------------
// Batched GEMM: C[g] = alpha * A[g] @ op(B[g]).  Group base A + (g>>dshA)*sA
// (dshA implements broadcast over L via shift).  TRANSB: B memory is (N x Kdim).
// grid = (N/16, M/16, G); all dims aligned.
// ---------------------------------------------------------------------------
template <bool TRANSB>
__global__ __launch_bounds__(32)
void bgemm_kernel(const float* __restrict__ A, long long sA, int dshA,
                  const float* __restrict__ B, long long sB, int dshB,
                  float* __restrict__ C, long long sC,
                  int N, int Kdim, float alpha) {
  int g = blockIdx.z;
  const float* Ag = A + (long long)(g >> dshA) * sA;
  const float* Bg = B + (long long)(g >> dshB) * sB;
  float* Cg = C + (long long)g * sC;
  int nBase = blockIdx.x << 4;
  int mBase = blockIdx.y << 4;
  int lane = threadIdx.x;
  v8f acc = {};
  for (int k0 = 0; k0 < Kdim; k0 += 32) {
    v16h af = load_a_frag(Ag, Kdim, mBase, k0, lane);
    v16h bf = TRANSB ? load_bt_frag(Bg, Kdim, k0, nBase, lane)
                     : load_b_frag(Bg, N, k0, nBase, lane);
    acc = __builtin_amdgcn_wmma_f32_16x16x32_f16(false, af, false, bf,
                                                 (short)0, acc, false, false);
  }
  int n = nBase + (lane & 15);
  int mOff = (lane & 16) ? 8 : 0;
#pragma unroll
  for (int r = 0; r < 8; ++r) {
    int m = mBase + r + mOff;
    Cg[(long long)m * N + n] = alpha * acc[r];
  }
}

// ---------------------------------------------------------------------------
// Row softmax over colsValid entries of a row with given stride; columns in
// [colsValid, stride) are written as 0 (kills padded attention keys).
// One wave per row; stride <= 128.
// ---------------------------------------------------------------------------
__global__ __launch_bounds__(32)
void softmax_rows(float* __restrict__ X, int rows, int colsValid, int stride) {
  int row = blockIdx.x;
  if (row >= rows) return;
  float* p = X + (long long)row * stride;
  int lane = threadIdx.x;
  float v[4];
  float mx = -1e30f;
#pragma unroll
  for (int i = 0; i < 4; ++i) {
    int c = lane + 32 * i;
    v[i] = (c < colsValid) ? p[c] : -1e30f;
    mx = fmaxf(mx, v[i]);
  }
#pragma unroll
  for (int off = 16; off > 0; off >>= 1) mx = fmaxf(mx, __shfl_xor(mx, off, 32));
  float sum = 0.0f;
#pragma unroll
  for (int i = 0; i < 4; ++i) {
    int c = lane + 32 * i;
    float e = (c < colsValid) ? __expf(v[i] - mx) : 0.0f;
    v[i] = e;
    sum += e;
  }
#pragma unroll
  for (int off = 16; off > 0; off >>= 1) sum += __shfl_xor(sum, off, 32);
  float inv = 1.0f / sum;
#pragma unroll
  for (int i = 0; i < 4; ++i) {
    int c = lane + 32 * i;
    if (c < stride) p[c] = (c < colsValid) ? v[i] * inv : 0.0f;
  }
}

// ---------------------------------------------------------------------------
// Pack kernels
// ---------------------------------------------------------------------------
// allobs_enc/map (B,64,64): rows 0-31 = nbrs, row 32 = obs, rows 33-63 = 0.
__global__ __launch_bounds__(256)
void pack_allobs(const float* __restrict__ nbrs, const float* __restrict__ obs,
                 const float* __restrict__ nbrmap, const float* __restrict__ obsmap,
                 float* __restrict__ aenc, float* __restrict__ amap, int total) {
  int i = blockIdx.x * blockDim.x + threadIdx.x;
  if (i >= total) return;
  int d = i & 63;
  int rest = i >> 6;
  int row = rest & 63;
  int b = rest >> 6;
  float e = 0.0f, m = 0.0f;
  if (row < 32) {
    long long s = ((long long)(b * 32 + row)) * 64 + d;
    e = nbrs[s];
    m = nbrmap[s];
  } else if (row == 32) {
    e = obs[b * 64 + d];
    m = obsmap[b * 64 + d];
  }
  aenc[i] = e;
  amap[i] = m;
}

// lane_comb (2048,128): [allobs_lane[g,32,:], social[g,32,:]]
__global__ __launch_bounds__(256)
void pack_lane_comb(const float* __restrict__ alane, const float* __restrict__ social,
                    float* __restrict__ out, int total) {
  int i = blockIdx.x * blockDim.x + threadIdx.x;
  if (i >= total) return;
  int c = i & 127;
  int g = i >> 7;
  long long base = ((long long)g * 64 + 32) * 64;
  out[i] = (c < 64) ? alane[base + c] : social[base + (c - 64)];
}

// traj_comb (32768,256): row (b, t=l*16+p) = [obs_part[b,l], inter_part[b,l], allpreds[b,t]]
__global__ __launch_bounds__(256)
void pack_traj_comb(const float* __restrict__ alane, const float* __restrict__ social,
                    const float* __restrict__ allpreds, float* __restrict__ out,
                    int total) {
  int i = blockIdx.x * blockDim.x + threadIdx.x;
  if (i >= total) return;
  int c = i & 255;
  int r = i >> 8;            // r = b*128 + t
  int b = r >> 7;
  int t = r & 127;
  int l = t >> 4;
  int g = b * 8 + l;
  long long lbase = ((long long)g * 64 + 32) * 64;
  float v;
  if (c < 64)       v = alane[lbase + c];
  else if (c < 128) v = social[lbase + (c - 64)];
  else              v = allpreds[(long long)r * 128 + (c - 128)];
  out[i] = v;
}

// Final decoder layer (N=1): out[m] = H[m,:] . w + b   (one wave per row)
__global__ __launch_bounds__(32)
void dot_out(const float* __restrict__ H, const float* __restrict__ w,
             const float* __restrict__ b, float* __restrict__ out, int Kdim) {
  int m = blockIdx.x;
  int lane = threadIdx.x;
  const float* h = H + (long long)m * Kdim;
  float s = 0.0f;
  for (int k = lane; k < Kdim; k += 32) s += h[k] * w[k];
#pragma unroll
  for (int off = 16; off > 0; off >>= 1) s += __shfl_xor(s, off, 32);
  if (lane == 0) out[m] = s + b[0];
}

// ---------------------------------------------------------------------------
// Host orchestration
// ---------------------------------------------------------------------------
extern "C" void kernel_launch(void* const* d_in, const int* in_sizes, int n_in,
                              void* d_out, int out_size, void* d_ws, size_t ws_size,
                              hipStream_t stream) {
  (void)in_sizes; (void)n_in; (void)out_size; (void)ws_size;

  const float* obs  = (const float*)d_in[0];   // (256,64)
  const float* cls  = (const float*)d_in[1];   // (256,8,64,64)
  const float* nbrs = (const float*)d_in[2];   // (256,32,64)
  const float* pred = (const float*)d_in[3];   // (256,8,16,128)
  const float* Wcls = (const float*)d_in[4];  const float* bcls = (const float*)d_in[5];
  const float* Wobs = (const float*)d_in[6];  const float* bobs = (const float*)d_in[7];
  const float* Wnbr = (const float*)d_in[8];  const float* bnbr = (const float*)d_in[9];
  const float* Wprd = (const float*)d_in[10]; const float* bprd = (const float*)d_in[11];
  const float* Wsoc = (const float*)d_in[12]; const float* bsoc = (const float*)d_in[13];
  const float* Wall = (const float*)d_in[14]; const float* ball = (const float*)d_in[15];
  const float* Wo_l2a = (const float*)d_in[16];
  const float* Wo_a2a = (const float*)d_in[17];
  const float* Wo_l2f = (const float*)d_in[18];
  const float* Wo_f2f = (const float*)d_in[19];
  const float* Ld1W = (const float*)d_in[20]; const float* Ld1b = (const float*)d_in[21];
  const float* Ld2W = (const float*)d_in[22]; const float* Ld2b = (const float*)d_in[23];
  const float* Ld3W = (const float*)d_in[24]; const float* Ld3b = (const float*)d_in[25];
  const float* Td1W = (const float*)d_in[26]; const float* Td1b = (const float*)d_in[27];
  const float* Td2W = (const float*)d_in[28]; const float* Td2b = (const float*)d_in[29];
  const float* Td3W = (const float*)d_in[30]; const float* Td3b = (const float*)d_in[31];

  float* ws = (float*)d_ws;
  float* out = (float*)d_out;   // [0,2048) lane_scores, [2048,34816) traj_scores

  // workspace layout (floats); agent dim padded 33 -> 64
  const size_t CLS_MAP   = 0;                    // 8388608 (B*L*64*64); later TRAJ_COMB
  const size_t S1        = CLS_MAP + 8388608;    // 8388608 (2048*64*64); later TRAJ_H1
  const size_t AV1       = S1 + 8388608;         // 8388608;             later TRAJ_H2
  const size_t ALANE     = AV1 + 8388608;        // 8388608
  const size_t SOCMAP    = ALANE + 8388608;      // 8388608
  const size_t S2        = SOCMAP + 8388608;     // 8388608
  const size_t AV2       = S2 + 8388608;         // 8388608
  const size_t SOCIAL    = AV2 + 8388608;        // 8388608
  const size_t PREDMAP   = SOCIAL + 8388608;     // 2097152 (32768*64)
  const size_t S3        = PREDMAP + 2097152;    // 2097152
  const size_t AV3       = S3 + 2097152;         // 2097152
  const size_t ALLPMAP   = AV3 + 2097152;        // 2097152
  const size_t AENC      = ALLPMAP + 2097152;    // 1048576 (256*64*64)
  const size_t AMAP      = AENC + 1048576;       // 1048576
  const size_t PRED_LANE = AMAP + 1048576;       // 4194304 (32768*128)
  const size_t S4        = PRED_LANE + 4194304;  // 4194304 (256*128*128)
  const size_t AV4       = S4 + 4194304;         // 4194304
  const size_t ALLPREDS  = AV4 + 4194304;        // 4194304
  const size_t LANE_COMB = ALLPREDS + 4194304;   // 262144 (2048*128)
  const size_t LANE_H1   = LANE_COMB + 262144;   // 262144
  const size_t LANE_H2   = LANE_H1 + 262144;     // total 95158272 floats (~381 MB)
  const size_t TRAJ_COMB = CLS_MAP;              // cls_map dead after s3
  const size_t TRAJ_H1   = S1;                   // s1 dead after av1
  const size_t TRAJ_H2   = AV1;                  // av1 dead after allobs_lane

  dim3 wv(32);

  // 1) projections (nbr/obs maps staged in not-yet-live AV2 / SOCIAL regions)
  gemm_kernel<0,0><<<dim3(4, 8192), wv, 0, stream>>>(cls,  Wcls, bcls, nullptr, ws + CLS_MAP, 64, 64);
  gemm_kernel<0,0><<<dim3(4, 512),  wv, 0, stream>>>(nbrs, Wnbr, bnbr, nullptr, ws + AV2,     64, 64);
  gemm_kernel<0,0><<<dim3(4, 16),   wv, 0, stream>>>(obs,  Wobs, bobs, nullptr, ws + SOCIAL,  64, 64);
  pack_allobs<<<dim3(1048576 / 256), dim3(256), 0, stream>>>(nbrs, obs, ws + AV2, ws + SOCIAL,
                                                             ws + AENC, ws + AMAP, 1048576);

  // 2) lane-to-agents attention (q rows padded to 64; pad q rows are zero)
  bgemm_kernel<true ><<<dim3(4, 4, 2048), wv, 0, stream>>>(ws + AMAP, 4096, 3, ws + CLS_MAP, 4096, 0,
                                                           ws + S1, 4096, 64, 64, 0.125f);
  softmax_rows<<<dim3(131072), wv, 0, stream>>>(ws + S1, 131072, 64, 64);
  bgemm_kernel<false><<<dim3(4, 4, 2048), wv, 0, stream>>>(ws + S1, 4096, 0, cls, 4096, 0,
                                                           ws + AV1, 4096, 64, 64, 1.0f);
  // allobs_lane = allobs_enc_l + av1 @ Wo_l2a (residual broadcast over L)
  gemm_kernel<0,2><<<dim3(4, 8192), wv, 0, stream>>>(ws + AV1, Wo_l2a, nullptr, ws + AENC, ws + ALANE, 64, 64);

  // 3) agent-to-agent attention (33 valid keys of 64; softmax masks + zeroes pads)
  gemm_kernel<0,0><<<dim3(4, 8192), wv, 0, stream>>>(ws + ALANE, Wsoc, bsoc, nullptr, ws + SOCMAP, 64, 64);
  bgemm_kernel<true ><<<dim3(4, 4, 2048), wv, 0, stream>>>(ws + SOCMAP, 4096, 0, ws + SOCMAP, 4096, 0,
                                                           ws + S2, 4096, 64, 64, 0.125f);
  softmax_rows<<<dim3(131072), wv, 0, stream>>>(ws + S2, 131072, 33, 64);
  bgemm_kernel<false><<<dim3(4, 4, 2048), wv, 0, stream>>>(ws + S2, 4096, 0, ws + ALANE, 4096, 0,
                                                           ws + AV2, 4096, 64, 64, 1.0f);
  gemm_kernel<0,1><<<dim3(4, 8192), wv, 0, stream>>>(ws + AV2, Wo_a2a, nullptr, ws + ALANE, ws + SOCIAL, 64, 64);

  // 4) lane-to-future attention
  gemm_kernel<0,0><<<dim3(4, 2048), wv, 0, stream>>>(pred, Wprd, bprd, nullptr, ws + PREDMAP, 64, 128);
  bgemm_kernel<true ><<<dim3(4, 1, 2048), wv, 0, stream>>>(ws + PREDMAP, 1024, 0, ws + CLS_MAP, 4096, 0,
                                                           ws + S3, 1024, 64, 64, 0.125f);
  softmax_rows<<<dim3(32768), wv, 0, stream>>>(ws + S3, 32768, 64, 64);
  bgemm_kernel<false><<<dim3(4, 1, 2048), wv, 0, stream>>>(ws + S3, 1024, 0, cls, 4096, 0,
                                                           ws + AV3, 1024, 64, 64, 1.0f);
  gemm_kernel<0,1><<<dim3(8, 2048), wv, 0, stream>>>(ws + AV3, Wo_l2f, nullptr, pred, ws + PRED_LANE, 128, 64);

  // 5) future-to-future attention (128x128 per batch)
  gemm_kernel<0,0><<<dim3(4, 2048), wv, 0, stream>>>(ws + PRED_LANE, Wall, ball, nullptr, ws + ALLPMAP, 64, 128);
  bgemm_kernel<true ><<<dim3(8, 8, 256), wv, 0, stream>>>(ws + ALLPMAP, 8192, 0, ws + ALLPMAP, 8192, 0,
                                                          ws + S4, 16384, 128, 64, 0.125f);
  softmax_rows<<<dim3(32768), wv, 0, stream>>>(ws + S4, 32768, 128, 128);
  bgemm_kernel<false><<<dim3(8, 8, 256), wv, 0, stream>>>(ws + S4, 16384, 0, ws + PRED_LANE, 16384, 0,
                                                          ws + AV4, 16384, 128, 128, 1.0f);
  gemm_kernel<0,1><<<dim3(8, 2048), wv, 0, stream>>>(ws + AV4, Wo_f2f, nullptr, ws + PRED_LANE, ws + ALLPREDS, 128, 128);

  // 6) pack decoder inputs
  pack_lane_comb<<<dim3(262144 / 256), dim3(256), 0, stream>>>(ws + ALANE, ws + SOCIAL, ws + LANE_COMB, 262144);
  pack_traj_comb<<<dim3(8388608 / 256), dim3(256), 0, stream>>>(ws + ALANE, ws + SOCIAL, ws + ALLPREDS,
                                                                ws + TRAJ_COMB, 8388608);

  // 7) lane decoder: 128 -> 128 -> 128 -> 1
  gemm_kernel<1,0><<<dim3(8, 128), wv, 0, stream>>>(ws + LANE_COMB, Ld1W, Ld1b, nullptr, ws + LANE_H1, 128, 128);
  gemm_kernel<1,0><<<dim3(8, 128), wv, 0, stream>>>(ws + LANE_H1,   Ld2W, Ld2b, nullptr, ws + LANE_H2, 128, 128);
  dot_out<<<dim3(2048), wv, 0, stream>>>(ws + LANE_H2, Ld3W, Ld3b, out, 128);

  // 8) traj decoder: 256 -> 256 -> 256 -> 1
  gemm_kernel<1,0><<<dim3(16, 2048), wv, 0, stream>>>(ws + TRAJ_COMB, Td1W, Td1b, nullptr, ws + TRAJ_H1, 256, 256);
  gemm_kernel<1,0><<<dim3(16, 2048), wv, 0, stream>>>(ws + TRAJ_H1,   Td2W, Td2b, nullptr, ws + TRAJ_H2, 256, 256);
  dot_out<<<dim3(32768), wv, 0, stream>>>(ws + TRAJ_H2, Td3W, Td3b, out + 2048, 256);
}
